// ConvLSTM1D_78529182040838
// MI455X (gfx1250) — compile-verified, matrix operand order checked by reading, not run on hardware
//
#include <hip/hip_runtime.h>
#include <hip/hip_bf16.h>

typedef __attribute__((ext_vector_type(16))) __bf16 v16bf;
typedef __attribute__((ext_vector_type(8)))  __bf16 v8bf;
typedef __attribute__((ext_vector_type(8)))  float  v8f;
typedef __attribute__((ext_vector_type(4)))  float  v4f;   // clang vector (not HIP struct)

#define BB     8
#define TT     128
#define CIN    64
#define COUT   64
#define HH     256
#define KCH    128            // CIN + COUT
#define MGATES 256            // 4*COUT
#define ROWP   136            // padded row length (bf16 elems) -> 272B, bank-conflict free
#define CAT_ROWS 258          // 256 positions + 1 zero pad each side

#define WLDS_ELEMS (3*MGATES*ROWP)   // 104448 bf16
#define CAT_ELEMS  (CAT_ROWS*ROWP)   // 35088  bf16
#define SMEM_BYTES (WLDS_ELEMS*2 + CAT_ELEMS*2 + MGATES*4)   // 280096 B < 320KB

__device__ __forceinline__ float sigmoidf_(float x) {
    return 1.0f / (1.0f + __expf(-x));
}
__device__ __forceinline__ float tanhf_(float x) {
    x = fminf(fmaxf(x, -15.0f), 15.0f);
    float e = __expf(2.0f * x);
    return (e - 1.0f) / (e + 1.0f);
}

// one WMMA operand fragment = two contiguous 16B chunks at k and k+16 elems
__device__ __forceinline__ v16bf load_frag(const __bf16* p) {
    v8bf lo = *(const v8bf*)p;
    v8bf hi = *(const v8bf*)(p + 16);
    return __builtin_shufflevector(lo, hi,
        0,1,2,3,4,5,6,7,8,9,10,11,12,13,14,15);
}

__global__ void __launch_bounds__(512, 1)
convlstm_wmma_kernel(const float* __restrict__ x,
                     const float* __restrict__ h0,
                     const float* __restrict__ c0,
                     const float* __restrict__ conv_w,
                     const float* __restrict__ conv_b,
                     const float* __restrict__ Wci,
                     const float* __restrict__ Wcf,
                     const float* __restrict__ Wco,
                     float* __restrict__ hs,
                     float* __restrict__ cs)
{
    extern __shared__ char smem[];
    __bf16* wlds = (__bf16*)smem;                                  // [3][256][ROWP]
    __bf16* catm = (__bf16*)(smem + WLDS_ELEMS*2);                 // [258][ROWP]
    float*  bias = (float*)(smem + WLDS_ELEMS*2 + CAT_ELEMS*2);    // [256]

    const int b    = blockIdx.x;
    const int tid  = threadIdx.x;
    const int w    = tid >> 5;      // wave id 0..15  -> owns positions 16w..16w+15
    const int lane = tid & 31;
    const int lhi  = lane >> 4;
    const int ln   = lane & 15;
    const int pcol = 16*w + ln;     // this lane's output position (N index)

    // ---------------- init ----------------
    for (int e = tid; e < CAT_ELEMS/2; e += 512) ((unsigned int*)catm)[e] = 0u;
    if (tid < MGATES) bias[tid] = conv_b[tid];

    // weights fp32 -> bf16 LDS, layout [tap][gate_row][in_ch] (row padded to ROWP)
    for (int e = tid; e < MGATES*KCH; e += 512) {
        int g = e >> 7, i = e & 127;
        const float* src = conv_w + e*3;
        wlds[(0*MGATES + g)*ROWP + i] = (__bf16)src[0];
        wlds[(1*MGATES + g)*ROWP + i] = (__bf16)src[1];
        wlds[(2*MGATES + g)*ROWP + i] = (__bf16)src[2];
    }
    // h0 -> cat cols [CIN, CIN+COUT)
    for (int e = tid; e < COUT*HH; e += 512) {
        int ch = e >> 8, p = e & 255;
        catm[(p+1)*ROWP + CIN + ch] = (__bf16)h0[(b*COUT + ch)*HH + p];
    }
    // c state in registers (WMMA C/D layout: VGPR r, M = r + 8*lhi, N = ln)
    float c_reg[4][8];
    #pragma unroll
    for (int j = 0; j < 4; ++j)
        #pragma unroll
        for (int r = 0; r < 8; ++r) {
            int ch = 16*j + 8*lhi + r;
            c_reg[j][r] = c0[(b*COUT + ch)*HH + pcol];
        }

    // register prefetch of x_0 (streamed once -> nontemporal)
    v4f xbuf[8];
    {
        const float* xn = x + ((b*TT + 0)*CIN)*HH;
        #pragma unroll
        for (int i = 0; i < 8; ++i)
            xbuf[i] = __builtin_nontemporal_load(
                          (const v4f*)(xn + ((tid + i*512) << 2)));
    }
    __syncthreads();

    const v8f vzero = {0.f,0.f,0.f,0.f,0.f,0.f,0.f,0.f};

    for (int t = 0; t < TT; ++t) {
        // ---- stage x_t from registers into cat cols [0, CIN) as bf16 ----
        #pragma unroll
        for (int i = 0; i < 8; ++i) {
            int f = (tid + i*512) << 2;
            int ch = f >> 8, p = f & 255;      // p multiple of 4
            v4f v = xbuf[i];
            catm[(p+1)*ROWP + ch] = (__bf16)v[0];
            catm[(p+2)*ROWP + ch] = (__bf16)v[1];
            catm[(p+3)*ROWP + ch] = (__bf16)v[2];
            catm[(p+4)*ROWP + ch] = (__bf16)v[3];
        }
        __syncthreads();   // x staged AND previous step's h writes visible

        // ---- issue prefetch of x_{t+1}; latency hidden under GEMM+epilogue ----
        if (t + 1 < TT) {
            const float* xn = x + ((b*TT + (t+1))*CIN)*HH;
            #pragma unroll
            for (int i = 0; i < 8; ++i)
                xbuf[i] = __builtin_nontemporal_load(
                              (const v4f*)(xn + ((tid + i*512) << 2)));
        }

        // ---- GEMM: gates[256 x 256] = sum_tap W_tap[256x128] x cat_shift[128x256]
        v8f acc[4][4];
        #pragma unroll
        for (int j = 0; j < 4; ++j)
            #pragma unroll
            for (int g = 0; g < 4; ++g)
                acc[j][g] = vzero;

        for (int tap = 0; tap < 3; ++tap) {
            const __bf16* wt   = wlds + tap*MGATES*ROWP;
            const int     prow = 16*w + tap + ln;      // padded row this lane reads
            #pragma unroll
            for (int kb = 0; kb < 4; ++kb) {
                const int k0 = kb*32 + lhi*8;
                // B fragment (K x N = 32 x 16): lane = column(position)
                v16bf Bf = load_frag(catm + prow*ROWP + k0);

                // rotating 4-deep window of A fragments keeps ~8 b128 LDS
                // loads in flight under the WMMAs (avoids s_wait_dscnt 0 per op)
                v16bf Aq[4];
                #pragma unroll
                for (int i = 0; i < 4; ++i)
                    Aq[i] = load_frag(wt + (i*16 + ln)*ROWP + k0);

                #pragma unroll
                for (int mt = 0; mt < 16; ++mt) {
                    const int j = mt & 3, g = mt >> 2;   // gate = mt/4, ch-tile = mt%4
                    acc[j][g] = __builtin_amdgcn_wmma_f32_16x16x32_bf16(
                        false, Aq[mt & 3], false, Bf, (short)0,
                        acc[j][g], false, false);
                    if (mt + 4 < 16)
                        Aq[mt & 3] = load_frag(wt + ((mt+4)*16 + ln)*ROWP + k0);
                }
            }
        }

        // ---- LSTM pointwise epilogue + global stores (streamed -> nontemporal)
        float hreg[4][8];
        #pragma unroll
        for (int j = 0; j < 4; ++j) {
            const int chbase = 16*j + 8*lhi;
            #pragma unroll
            for (int r = 0; r < 8; ++r) {
                const int ch   = chbase + r;
                const int wcix = ch*HH + pcol;
                float cprev = c_reg[j][r];
                float iv = sigmoidf_(acc[j][0][r] + bias[0*COUT + ch] + Wci[wcix]*cprev);
                float fv = sigmoidf_(acc[j][1][r] + bias[1*COUT + ch] + Wcf[wcix]*cprev);
                float gv = tanhf_   (acc[j][2][r] + bias[2*COUT + ch]);
                float cn = fv*cprev + iv*gv;
                float ov = sigmoidf_(acc[j][3][r] + bias[3*COUT + ch] + Wco[wcix]*cn);
                float hn = ov*tanhf_(cn);
                c_reg[j][r] = cn;
                hreg[j][r]  = hn;
                const int oidx = ((b*TT + t)*COUT + ch)*HH + pcol;
                __builtin_nontemporal_store(hn, &hs[oidx]);
                __builtin_nontemporal_store(cn, &cs[oidx]);
            }
        }
        __syncthreads();   // all GEMM reads of cat done before h overwrite

        // ---- write h_new back into cat cols [CIN, CIN+COUT) for next step ----
        #pragma unroll
        for (int j = 0; j < 4; ++j) {
            __bf16* dst = catm + (pcol+1)*ROWP + CIN + 16*j + 8*lhi;
            #pragma unroll
            for (int r = 0; r < 8; ++r) dst[r] = (__bf16)hreg[j][r];
        }
        // next iteration's first barrier fences these writes vs. GEMM reads
    }
}

extern "C" void kernel_launch(void* const* d_in, const int* in_sizes, int n_in,
                              void* d_out, int out_size, void* d_ws, size_t ws_size,
                              hipStream_t stream) {
    (void)in_sizes; (void)n_in; (void)out_size; (void)d_ws; (void)ws_size;
    const float* x      = (const float*)d_in[0];
    const float* h0     = (const float*)d_in[1];
    const float* c0     = (const float*)d_in[2];
    const float* conv_w = (const float*)d_in[3];
    const float* conv_b = (const float*)d_in[4];
    const float* Wci    = (const float*)d_in[5];
    const float* Wcf    = (const float*)d_in[6];
    const float* Wco    = (const float*)d_in[7];
    float* hs = (float*)d_out;
    float* cs = hs + (size_t)BB*TT*COUT*HH;   // tuple (hs, cs) concatenated

    // allow 280KB dynamic LDS (gfx1250 WGP has 320KB)
    (void)hipFuncSetAttribute((const void*)convlstm_wmma_kernel,
                              hipFuncAttributeMaxDynamicSharedMemorySize,
                              SMEM_BYTES);

    convlstm_wmma_kernel<<<dim3(BB), dim3(512), SMEM_BYTES, stream>>>(
        x, h0, c0, conv_w, conv_b, Wci, Wcf, Wco, hs, cs);
}
